// CrossModalAttention_35399120454140
// MI455X (gfx1250) — matrix operand order repaired and checked by reading, not compiled, not run on hardware
//
#include <hip/hip_runtime.h>
#include <hip/hip_bf16.h>

// CDNA5 (gfx1250, wave32) fused cross-modal attention.
// [gemm Q/K/V -> bf16] -> [fused attention: head loop in-block, LDS scores,
// attn-mean accumulated] -> [fused outproj + bias + residual + LayerNorm].
// All matmuls: v_wmma_f32_16x16x32_bf16. LDS tiles are laid out so each
// lane's WMMA fragment is contiguous -> ds_load_b128 pairs per fragment.

typedef __bf16 bf16_t;
typedef __attribute__((ext_vector_type(16))) __bf16 v16bf;
typedef __attribute__((ext_vector_type(8)))  __bf16 v8bf;
typedef __attribute__((ext_vector_type(4)))  __bf16 v4bf;
typedef __attribute__((ext_vector_type(8)))  float  v8f;
typedef __attribute__((ext_vector_type(4)))  float  v4f;

#define NUM_HEADS 16
#define HEAD_DIM  64
#define INNER     1024
#define CDIM      1024
#define SDIM      768
#define LN_EPS    1e-5f

// Build a 16-element bf16 fragment from two contiguous 16-byte LDS runs.
__device__ __forceinline__ v16bf frag2(const bf16_t* p0, const bf16_t* p1) {
    v8bf lo = *(const v8bf*)p0;
    v8bf hi = *(const v8bf*)p1;
    return __builtin_shufflevector(lo, hi, 0,1,2,3,4,5,6,7,8,9,10,11,12,13,14,15);
}
// A-fragment (16x32 tile, row-major, k contiguous): lane covers
// k = {half*8 .. +8} and {16+half*8 .. +8}   (ISA 7.12.2)
__device__ __forceinline__ v16bf afrag(const bf16_t* row, int kk, int half) {
    return frag2(row + kk + half * 8, row + kk + 16 + half * 8);
}
// B-fragment (32x16 tile stored transposed [n][k], k contiguous): lane covers
// k = {half*8 .. +8} and {16+half*8 .. +8}
__device__ __forceinline__ v16bf bfrag(const bf16_t* rowN, int kk, int half) {
    return frag2(rowN + kk + half * 8, rowN + kk + 16 + half * 8);
}

__device__ __forceinline__ float wave_max(float v) {
    #pragma unroll
    for (int off = 16; off > 0; off >>= 1) v = fmaxf(v, __shfl_xor(v, off, 32));
    return v;
}
__device__ __forceinline__ float wave_sum(float v) {
    #pragma unroll
    for (int off = 16; off > 0; off >>= 1) v += __shfl_xor(v, off, 32);
    return v;
}

// ---------------------------------------------------------------------------
// Kernel 1: Out_bf16[M,N] = bf16( (A_f32[M,K] * alpha) @ W_f32[K,N] )
// 256 threads = 8 waves (2x4); block tile 128x128; K step 32.
// Each wave: 4 A-frags x 2 B-frags -> 8 WMMAs per 12 fragment loads.
// ---------------------------------------------------------------------------
__global__ __launch_bounds__(256)
void gemm_bf16_kernel(const float* __restrict__ A, const float* __restrict__ W,
                      bf16_t* __restrict__ Out, int M, int K, int N, float alpha)
{
    __shared__ __align__(16) bf16_t As[128][32];   // [m][k] row-major, 8 KB
    __shared__ __align__(16) bf16_t Bt[128][40];   // [n][k] transposed, 10 KB

    const int tid  = threadIdx.x;
    const int lane = tid & 31;
    const int wave = tid >> 5;      // 0..7
    const int half = lane >> 4;
    const int l16  = lane & 15;
    const int wr   = wave >> 2;     // 0..1 : 64-row group
    const int wc   = wave & 3;      // 0..3 : 32-col group
    const int bn   = blockIdx.x * 128;
    const int bm   = blockIdx.y * 128;

    v8f acc[4][2] = {};

    for (int k0 = 0; k0 < K; k0 += 32) {
        // A tile: 128x32 f32 -> bf16, float4 reads, bf16x4 stores
        for (int i = tid; i < 1024; i += 256) {
            int r = i >> 3, c4 = (i & 7) * 4;
            v4f a = *(const v4f*)&A[(size_t)(bm + r) * K + k0 + c4];
            v4bf o;
            #pragma unroll
            for (int j = 0; j < 4; ++j) o[j] = (bf16_t)(a[j] * alpha);
            *(v4bf*)&As[r][c4] = o;
        }
        // W tile: 32x128, coalesced float4 reads, transposed bf16 stores
        for (int i = tid; i < 1024; i += 256) {
            int r = i >> 5, c4 = (i & 31) * 4;
            v4f w = *(const v4f*)&W[(size_t)(k0 + r) * N + bn + c4];
            #pragma unroll
            for (int j = 0; j < 4; ++j) Bt[c4 + j][r] = (bf16_t)w[j];
        }
        __syncthreads();

        v16bf af[4], bf[2];
        #pragma unroll
        for (int mi = 0; mi < 4; ++mi)
            af[mi] = afrag(&As[wr * 64 + mi * 16 + l16][0], 0, half);
        #pragma unroll
        for (int ni = 0; ni < 2; ++ni)
            bf[ni] = bfrag(&Bt[wc * 32 + ni * 16 + l16][0], 0, half);

        #pragma unroll
        for (int mi = 0; mi < 4; ++mi)
            #pragma unroll
            for (int ni = 0; ni < 2; ++ni)
                acc[mi][ni] = __builtin_amdgcn_wmma_f32_16x16x32_bf16(
                    false, af[mi], false, bf[ni], (short)0, acc[mi][ni],
                    false, false);
        __syncthreads();
    }

    #pragma unroll
    for (int mi = 0; mi < 4; ++mi)
        #pragma unroll
        for (int ni = 0; ni < 2; ++ni)
            #pragma unroll
            for (int r = 0; r < 8; ++r) {
                int m = bm + wr * 64 + mi * 16 + r + half * 8;
                int n = bn + wc * 32 + ni * 16 + l16;
                Out[(size_t)m * N + n] = (bf16_t)acc[mi][ni][r];
            }
}

// ---------------------------------------------------------------------------
// Kernel 2: fused attention for one (batch, 32-query tile); head loop inside.
// 256 threads = 8 waves; per stage each wave owns one 16x16 tile (2q x 4s|d).
// ---------------------------------------------------------------------------
__global__ __launch_bounds__(256)
void attn_kernel(const bf16_t* __restrict__ Qb,   // [B*NC, INNER]
                 const bf16_t* __restrict__ Kb,   // [B*NS, INNER]
                 const bf16_t* __restrict__ Vb,   // [B*NS, INNER]
                 bf16_t* __restrict__ AVb,        // [B*NC, INNER] (aliases Qb)
                 float* __restrict__ amean,       // [B*NC, NS]
                 int NC, int NS)
{
    __shared__ __align__(16) bf16_t qs[32][HEAD_DIM];   // [q][d], 4 KB
    __shared__ __align__(16) bf16_t kvs[64][72];        // K:[s][d] / Vt:[d][s]
    __shared__ __align__(16) float  ss[32][1024];       // scores / bf16 probs

    const int tid  = threadIdx.x;
    const int lane = tid & 31;
    const int wave = tid >> 5;                    // 0..7
    const int half = lane >> 4;
    const int l16  = lane & 15;

    const int bq = blockIdx.x;
    const int b  = bq / (NC / 32);
    const int q0 = (bq % (NC / 32)) * 32;
    const size_t qrow0 = (size_t)b * NC + q0;
    const size_t krow0 = (size_t)b * NS;

    const int qi = wave >> 2;                     // query 16-tile
    const int ti = wave & 3;                      // key/dim 16-tile

    for (int h = 0; h < NUM_HEADS; ++h) {
        // ---- Q tile (bf16x8 loads) ------------------------------------
        for (int i = tid; i < 256; i += 256) {
            int r = i >> 3, c8 = (i & 7) * 8;
            *(v8bf*)&qs[r][c8] =
                *(const v8bf*)&Qb[(qrow0 + r) * INNER + h * HEAD_DIM + c8];
        }
        __syncthreads();

        // ---- scores S[32 x NS] ----------------------------------------
        for (int s0 = 0; s0 < NS; s0 += 64) {
            for (int i = tid; i < 512; i += 256) {       // K tile [s][d]
                int r = i >> 3, c8 = (i & 7) * 8;
                *(v8bf*)&kvs[r][c8] =
                    *(const v8bf*)&Kb[(krow0 + s0 + r) * INNER + h * HEAD_DIM + c8];
            }
            __syncthreads();

            v8f c = {};
            #pragma unroll
            for (int kk = 0; kk < HEAD_DIM; kk += 32) {
                v16bf af = afrag(&qs[qi * 16 + l16][0], kk, half);
                v16bf bf = bfrag(&kvs[ti * 16 + l16][0], kk, half); // B[d][s]=K[s][d]
                c = __builtin_amdgcn_wmma_f32_16x16x32_bf16(
                    false, af, false, bf, (short)0, c, false, false);
            }
            #pragma unroll
            for (int r = 0; r < 8; ++r)
                ss[qi * 16 + r + half * 8][s0 + ti * 16 + l16] = c[r];
            __syncthreads();
        }

        // ---- softmax (4 rows/wave) + attn-mean + in-place bf16 probs --
        for (int rr = 0; rr < 4; ++rr) {
            const int r = wave * 4 + rr;
            float* row = &ss[r][0];
            float m = -1e30f;
            for (int j = lane; j < NS; j += 32) m = fmaxf(m, row[j]);
            m = wave_max(m);
            float s = 0.f;
            for (int j = lane; j < NS; j += 32) s += __expf(row[j] - m);
            s = wave_sum(s);
            const float inv = 1.f / s;
            bf16_t* prow = (bf16_t*)row;            // write byte 2j <= read 4j
            float* arow = amean + (qrow0 + r) * NS;
            for (int j = lane; j < NS; j += 32) {
                float p = __expf(row[j] - m) * inv;
                float contrib = p * (1.0f / NUM_HEADS);
                if (h == 0) arow[j] = contrib; else arow[j] += contrib;
                prow[j] = (bf16_t)p;
            }
        }
        __syncthreads();

        // ---- AV = P @ V -----------------------------------------------
        v8f avc = {};
        for (int s0 = 0; s0 < NS; s0 += 64) {
            // V tile transposed into kvs as [d][s]
            for (int i = tid; i < 512; i += 256) {
                int s = i >> 3, d0 = (i & 7) * 8;
                v8bf v =
                    *(const v8bf*)&Vb[(krow0 + s0 + s) * INNER + h * HEAD_DIM + d0];
                #pragma unroll
                for (int j = 0; j < 8; ++j) kvs[d0 + j][s] = v[j];
            }
            __syncthreads();
            const bf16_t* prow = (const bf16_t*)&ss[qi * 16 + l16][0];
            #pragma unroll
            for (int kk = 0; kk < 64; kk += 32) {
                v16bf af = afrag(prow, s0 + kk, half);
                v16bf bf = bfrag(&kvs[ti * 16 + l16][0], kk, half); // Vt[d][s]
                avc = __builtin_amdgcn_wmma_f32_16x16x32_bf16(
                    false, af, false, bf, (short)0, avc, false, false);
            }
            __syncthreads();
        }
        #pragma unroll
        for (int r = 0; r < 8; ++r) {
            size_t m = qrow0 + qi * 16 + r + half * 8;
            AVb[m * INNER + h * HEAD_DIM + ti * 16 + l16] = (bf16_t)avc[r];
        }
        __syncthreads();
    }
}

// ---------------------------------------------------------------------------
// Kernel 3: out = LayerNorm(content + AV @ Wo + bo) * gamma + beta.
// 256 threads = 8 waves; 16 full rows per block (LN fused, no extra pass).
// ---------------------------------------------------------------------------
__global__ __launch_bounds__(256)
void outproj_ln_kernel(const bf16_t* __restrict__ AVb,    // [M, INNER]
                       const float* __restrict__ Wo,      // [INNER, CDIM]
                       const float* __restrict__ bo,
                       const float* __restrict__ content, // [M, CDIM]
                       const float* __restrict__ gamma,
                       const float* __restrict__ beta,
                       float* __restrict__ outp)          // [M, CDIM]
{
    __shared__ __align__(16) bf16_t As[16][32];   // [m][k]
    __shared__ __align__(16) bf16_t Wt[1024][40]; // [n][k] transposed, 80 KB
    float* xs = (float*)&Wt[0][0];                // 16*1024 f32 = 64 KB reuse

    const int tid  = threadIdx.x;
    const int lane = tid & 31;
    const int wave = tid >> 5;                    // 0..7
    const int half = lane >> 4;
    const int l16  = lane & 15;
    const int row0 = blockIdx.x * 16;

    v8f acc[8] = {};

    for (int k0 = 0; k0 < INNER; k0 += 32) {
        if (tid < 64) {                           // A tile: 64 chunks of 8
            int r = tid >> 2, c8 = (tid & 3) * 8;
            *(v8bf*)&As[r][c8] =
                *(const v8bf*)&AVb[(size_t)(row0 + r) * INNER + k0 + c8];
        }
        // Wo tile 32x1024: coalesced float4 reads, transposed bf16 stores
        for (int i = tid; i < 8192; i += 256) {
            int r = i >> 8, c4 = (i & 255) * 4;
            v4f w = *(const v4f*)&Wo[(size_t)(k0 + r) * CDIM + c4];
            #pragma unroll
            for (int j = 0; j < 4; ++j) Wt[c4 + j][r] = (bf16_t)w[j];
        }
        __syncthreads();

        v16bf af = afrag(&As[l16][0], 0, half);
        #pragma unroll
        for (int t = 0; t < 8; ++t) {
            const int n0 = wave * 128 + t * 16;
            v16bf bf = bfrag(&Wt[n0 + l16][0], 0, half);
            acc[t] = __builtin_amdgcn_wmma_f32_16x16x32_bf16(
                false, af, false, bf, (short)0, acc[t], false, false);
        }
        __syncthreads();
    }

    // bias + residual into xs (Wt no longer needed)
    #pragma unroll
    for (int t = 0; t < 8; ++t) {
        const int n = wave * 128 + t * 16 + l16;
        #pragma unroll
        for (int r = 0; r < 8; ++r) {
            const int m = r + half * 8;
            xs[m * 1024 + n] =
                acc[t][r] + bo[n] + content[(size_t)(row0 + m) * CDIM + n];
        }
    }
    __syncthreads();

    // LayerNorm: 2 rows per wave
    for (int rr = 0; rr < 2; ++rr) {
        const int m = wave * 2 + rr;
        const float* row = &xs[m * 1024];
        float s = 0.f;
        for (int j = lane; j < CDIM; j += 32) s += row[j];
        const float mu = wave_sum(s) * (1.0f / CDIM);
        float v = 0.f;
        for (int j = lane; j < CDIM; j += 32) { float d = row[j] - mu; v += d * d; }
        const float inv = rsqrtf(wave_sum(v) * (1.0f / CDIM) + LN_EPS);
        for (int j = lane; j < CDIM; j += 32)
            outp[(size_t)(row0 + m) * CDIM + j] =
                (row[j] - mu) * inv * gamma[j] + beta[j];
    }
}

// ---------------------------------------------------------------------------
extern "C" void kernel_launch(void* const* d_in, const int* in_sizes, int n_in,
                              void* d_out, int out_size, void* d_ws, size_t ws_size,
                              hipStream_t stream)
{
    const int B = 4, NC = 4096, NS = 1024;
    const float* content = (const float*)d_in[0];
    const float* style   = (const float*)d_in[1];
    const float* Wq      = (const float*)d_in[2];
    const float* Wk      = (const float*)d_in[3];
    const float* Wv      = (const float*)d_in[4];
    const float* Wo      = (const float*)d_in[5];
    const float* bo      = (const float*)d_in[6];
    const float* gamma   = (const float*)d_in[7];
    const float* beta    = (const float*)d_in[8];

    float* outp  = (float*)d_out;                 // [B*NC, CDIM]
    float* amean = outp + (size_t)B * NC * CDIM;  // [B*NC, NS]

    bf16_t* Qb  = (bf16_t*)d_ws;
    bf16_t* Kb  = Qb + (size_t)B * NC * INNER;
    bf16_t* Vb  = Kb + (size_t)B * NS * INNER;
    bf16_t* AVb = Qb;  // safe alias: AV writes head-h cols strictly after all
                       // Q reads of those cols for that block's rows.

    const int Mq = B * NC;   // 16384
    const int Ms = B * NS;   // 4096

    gemm_bf16_kernel<<<dim3(INNER / 128, Mq / 128), 256, 0, stream>>>(
        content, Wq, Qb, Mq, CDIM, INNER, 0.125f);  // scale folded into Q
    gemm_bf16_kernel<<<dim3(INNER / 128, Ms / 128), 256, 0, stream>>>(
        style, Wk, Kb, Ms, SDIM, INNER, 1.0f);
    gemm_bf16_kernel<<<dim3(INNER / 128, Ms / 128), 256, 0, stream>>>(
        style, Wv, Vb, Ms, SDIM, INNER, 1.0f);

    attn_kernel<<<dim3(Mq / 32), 256, 0, stream>>>(
        Qb, Kb, Vb, AVb, amean, NC, NS);

    outproj_ln_kernel<<<dim3(Mq / 16), 256, 0, stream>>>(
        AVb, Wo, bo, content, gamma, beta, outp);
}
